// CAT_Transformer_32014686224751
// MI455X (gfx1250) — matrix operand order repaired and checked by reading, not compiled, not run on hardware
//
#include <hip/hip_runtime.h>
#include <hip/hip_bf16.h>
#include <math.h>
#include <stdint.h>

// ---------------- CDNA5 WMMA types ----------------
typedef __attribute__((ext_vector_type(16))) __bf16 v16bf;
typedef __attribute__((ext_vector_type(8)))  float  v8f;

// Model constants
#define BATCH_   2
#define KCH      128          // chunks per batch
#define LCH      17           // tokens per chunk (CS+1)
#define CS_      16
#define FD_      512
#define FNH_     8
#define DIM_     1024
#define NH_      16
#define SEQC     2177         // K*L + 1
#define FT_      (BATCH_*KCH*LCH)   // 4352 fine tokens
#define CT_      (BATCH_*SEQC)      // 4354 coarse tokens

// =====================================================================
// Fused transpose + f32->bf16 weight conversion:
//   W[Kd][N] f32  ->  WT[N][Kd] bf16     (32x32 tiles via LDS)
// Both global access patterns coalesced. Kd, N multiples of 32.
// =====================================================================
__global__ __launch_bounds__(256) void k_cvtT(const float* __restrict__ W,
                                              __bf16* __restrict__ WT,
                                              int Kd, int N) {
    __shared__ float tile[32][33];
    int n0 = blockIdx.x * 32, k0 = blockIdx.y * 32;
    int tx = threadIdx.x & 31, ty = threadIdx.x >> 5;     // ty 0..7
    for (int i = ty; i < 32; i += 8)
        tile[i][tx] = W[(long)(k0 + i) * N + n0 + tx];
    __syncthreads();
    for (int i = ty; i < 32; i += 8)
        WT[(long)(n0 + i) * Kd + k0 + tx] = (__bf16)tile[tx][i];
}

// =====================================================================
// Tiled WMMA GEMM: C[M,N] = A[M,K](f32, row stride lda) * W[K,N]
// where BwT is the pre-transposed bf16 weight [N][Kd].
//   mode 0: C = acc ; mode 1: C = acc + Aux ; mode 2: C = acc * silu(Aux)
// N % 128 == 0, Kd % 32 == 0 guaranteed by model dims.
// Double-buffered LDS; B tile staged with GLOBAL_LOAD_ASYNC_TO_LDS (DMA,
// ASYNCcnt), A tile staged via VGPRs with f32->bf16 conversion.
// =====================================================================
__global__ __launch_bounds__(256) void k_gemm_bf16(
    const float* __restrict__ A, long lda,
    const __bf16* __restrict__ BwT,
    float* __restrict__ C,
    const float* __restrict__ Aux,
    int M, int N, int Kd, int mode)
{
    __shared__ __bf16 As[2][128 * 32];   // [m][k]
    __shared__ __bf16 Bs[2][128 * 32];   // [n][k]

    const int tid  = threadIdx.x;
    const int lane = tid & 31;
    const int wv   = tid >> 5;        // 0..7
    const int wrow = wv >> 2;         // 0..1  -> 64-row slab
    const int wcol = wv & 3;          // 0..3  -> 32-col slab
    const int al   = lane & 15;
    const int h16  = lane >> 4;       // upper half-wave flag
    const int bm0  = blockIdx.y * 128;
    const int bn0  = blockIdx.x * 128;

    v8f acc[4][2];
#pragma unroll
    for (int t = 0; t < 4; ++t)
#pragma unroll
        for (int u = 0; u < 2; ++u)
#pragma unroll
            for (int j = 0; j < 8; ++j) acc[t][u][j] = 0.f;

    // staging geometry: each thread owns one 16-element half-row
    const int srow  = tid >> 1;       // 0..127 (A: m-row, B: n-row)
    const int shalf = tid & 1;        // which 16-elem half

    int agr = bm0 + srow; if (agr >= M) agr = M - 1;
    const float*  aBase = A   + (long)agr * lda + shalf * 16;
    const __bf16* bBase = BwT + (long)(bn0 + srow) * Kd + shalf * 16;

    // LDS byte addresses for the async DMA destination (low 32 bits of
    // the generic pointer hold the LDS offset)
    unsigned ldsB[2];
    ldsB[0] = (unsigned)(uintptr_t)&Bs[0][srow * 32 + shalf * 16];
    ldsB[1] = (unsigned)(uintptr_t)&Bs[1][srow * 32 + shalf * 16];

    // ---- prologue: stage k0 = 0 into buffer 0 ----
    {
        unsigned long long g = (unsigned long long)(uintptr_t)bBase;
#if defined(__HIP_DEVICE_COMPILE__)
        asm volatile(
            "global_load_async_to_lds_b128 %0, %1, off\n\t"
            "global_load_async_to_lds_b128 %0, %1, off offset:16"
            :: "v"(ldsB[0]), "v"(g) : "memory");
#endif
        float fr[16];
        ((float4*)fr)[0] = ((const float4*)aBase)[0];
        ((float4*)fr)[1] = ((const float4*)aBase)[1];
        ((float4*)fr)[2] = ((const float4*)aBase)[2];
        ((float4*)fr)[3] = ((const float4*)aBase)[3];
        __bf16 tb[16];
#pragma unroll
        for (int i = 0; i < 16; ++i) tb[i] = (__bf16)fr[i];
        uint4* dst = (uint4*)&As[0][srow * 32 + shalf * 16];
        dst[0] = ((uint4*)tb)[0];
        dst[1] = ((uint4*)tb)[1];
    }

    int cur = 0;
    for (int k0 = 0; k0 < Kd; k0 += 32) {
#if defined(__HIP_DEVICE_COMPILE__)
        asm volatile("s_wait_asynccnt 0x0" ::: "memory");
#endif
        __syncthreads();

        const bool hasNext = (k0 + 32) < Kd;
        float fr[16];
        if (hasNext) {
            // B tile for next step -> other LDS buffer via async DMA
            unsigned long long g = (unsigned long long)(uintptr_t)(bBase + k0 + 32);
#if defined(__HIP_DEVICE_COMPILE__)
            asm volatile(
                "global_load_async_to_lds_b128 %0, %1, off\n\t"
                "global_load_async_to_lds_b128 %0, %1, off offset:16"
                :: "v"(ldsB[cur ^ 1]), "v"(g) : "memory");
#endif
            // A tile for next step -> VGPRs (converted after compute)
            const float* ap = aBase + k0 + 32;
            ((float4*)fr)[0] = ((const float4*)ap)[0];
            ((float4*)fr)[1] = ((const float4*)ap)[1];
            ((float4*)fr)[2] = ((const float4*)ap)[2];
            ((float4*)fr)[3] = ((const float4*)ap)[3];
        }

        // ---- fragments per ISA 16-bit layouts ----
        union FragA { v16bf v; uint4 q[2]; } fa[4];
        union FragB { v16bf v; uint4 q[2]; } fb[2];
#pragma unroll
        for (int t = 0; t < 4; ++t) {
            // lanes<16: K 0-7 then 16-23 ; lanes>=16: K 8-15 then 24-31
            const __bf16* p = &As[cur][(wrow * 64 + t * 16 + al) * 32 + h16 * 8];
            fa[t].q[0] = *(const uint4*)p;
            fa[t].q[1] = *(const uint4*)(p + 16);
        }
#pragma unroll
        for (int u = 0; u < 2; ++u) {
            // lanes<16: K 0-7 then 16-23 of col n ; lanes>=16: K 8-15 then 24-31
            const __bf16* p = &Bs[cur][(wcol * 32 + u * 16 + al) * 32 + h16 * 8];
            fb[u].q[0] = *(const uint4*)p;
            fb[u].q[1] = *(const uint4*)(p + 16);
        }
#if defined(__HIP_DEVICE_COMPILE__)
#pragma unroll
        for (int t = 0; t < 4; ++t)
#pragma unroll
            for (int u = 0; u < 2; ++u)
                acc[t][u] = __builtin_amdgcn_wmma_f32_16x16x32_bf16(
                    false, fa[t].v, false, fb[u].v,
                    (short)0, acc[t][u], false, false);
#endif
        if (hasNext) {
            __bf16 tb[16];
#pragma unroll
            for (int i = 0; i < 16; ++i) tb[i] = (__bf16)fr[i];
            uint4* dst = (uint4*)&As[cur ^ 1][srow * 32 + shalf * 16];
            dst[0] = ((uint4*)tb)[0];
            dst[1] = ((uint4*)tb)[1];
        }
        cur ^= 1;
    }

    // ---- epilogue (C layout: VGPR j -> row 16t+j(+8 upper half), col = lane&15) ----
#pragma unroll
    for (int t = 0; t < 4; ++t)
#pragma unroll
        for (int u = 0; u < 2; ++u)
#pragma unroll
            for (int j = 0; j < 8; ++j) {
                int row = bm0 + wrow * 64 + t * 16 + j + h16 * 8;
                int col = bn0 + wcol * 32 + u * 16 + al;
                if (row < M) {
                    long idx = (long)row * N + col;
                    float v = acc[t][u][j];
                    if (mode == 1) v += Aux[idx];
                    else if (mode == 2) { float a = Aux[idx]; v *= a / (1.f + __expf(-a)); }
                    C[idx] = v;
                }
            }
}

// =====================================================================
// RMSNorm: one block per row
// =====================================================================
__global__ __launch_bounds__(256) void k_rmsnorm(const float* __restrict__ x,
                                                 const float* __restrict__ w,
                                                 float* __restrict__ y, int D) {
    __shared__ float red[256];
    int row = blockIdx.x, tid = threadIdx.x;
    const float* xr = x + (long)row * D;
    float s = 0.f;
    for (int i = tid; i < D; i += 256) { float v = xr[i]; s += v * v; }
    red[tid] = s; __syncthreads();
    for (int o = 128; o > 0; o >>= 1) { if (tid < o) red[tid] += red[tid + o]; __syncthreads(); }
    float inv = rsqrtf(red[0] / (float)D + 1e-5f);
    for (int i = tid; i < D; i += 256) y[(long)row * D + i] = xr[i] * inv * w[i];
}

// =====================================================================
// RoPE in place on qkv [ntok, 3*nh*64]; mode 0 fine (pos = tok%17),
// mode 1 coarse-cat (pos = chunk + l, last token pos = 128)
// =====================================================================
__global__ __launch_bounds__(256) void k_rope(float* __restrict__ qkv,
                                              int ntok, int nh, int mode) {
    int per = 2 * nh * 32;
    long gid = (long)blockIdx.x * blockDim.x + threadIdx.x;
    if (gid >= (long)ntok * per) return;
    int tok = (int)(gid / per);
    int r   = (int)(gid % per);
    int mat = r / (nh * 32);
    int h   = (r / 32) % nh;
    int d   = r & 31;
    int pos;
    if (mode == 0) pos = tok % LCH;
    else { int s = tok % SEQC; pos = (s >= KCH * LCH) ? KCH : (s / LCH + s % LCH); }
    float theta = __powf(10000.f, -(float)(2 * d) / 64.f);
    float sn, c; __sincosf((float)pos * theta, &sn, &c);
    float* p = qkv + (long)tok * (3 * nh * 64) + mat * nh * 64 + h * 64 + d;
    float x1 = p[0], x2 = p[32];
    p[0]  = x1 * c - x2 * sn;
    p[32] = x2 * c + x1 * sn;
}

// =====================================================================
// Fine full (unmasked) attention: block per (seq,head), 64 threads
// qkv [seq*17, 1536], out [seq*17, 512]
// =====================================================================
__global__ __launch_bounds__(64) void k_attn_full(const float* __restrict__ qkv,
                                                  float* __restrict__ out) {
    __shared__ float qs[17][64], ks[17][64], vs[17][64], sc[17];
    int seq = blockIdx.x >> 3, h = blockIdx.x & 7, tid = threadIdx.x;
    for (int i = tid; i < 17 * 64; i += 64) {
        int l = i >> 6, d = i & 63;
        long base = (long)(seq * 17 + l) * 1536;
        qs[l][d] = qkv[base + h * 64 + d];
        ks[l][d] = qkv[base + 512 + h * 64 + d];
        vs[l][d] = qkv[base + 1024 + h * 64 + d];
    }
    __syncthreads();
    for (int l = 0; l < 17; ++l) {
        if (tid < 17) {
            float s = 0.f;
            for (int d = 0; d < 64; ++d) s += qs[l][d] * ks[tid][d];
            sc[tid] = s * 0.125f;
        }
        __syncthreads();
        if (tid == 0) {
            float m = -1e30f;
            for (int j = 0; j < 17; ++j) m = fmaxf(m, sc[j]);
            float sum = 0.f;
            for (int j = 0; j < 17; ++j) { sc[j] = __expf(sc[j] - m); sum += sc[j]; }
            float r = 1.f / sum;
            for (int j = 0; j < 17; ++j) sc[j] *= r;
        }
        __syncthreads();
        float o = 0.f;
        for (int j = 0; j < 17; ++j) o += sc[j] * vs[j][tid];
        out[(long)(seq * 17 + l) * 512 + h * 64 + tid] = o;
        __syncthreads();
    }
}

// =====================================================================
// Coarse CAT attention: grid (K+1, NH, B), 64 threads
// qkv [b*2177+s, 3072]; out [b*2177+s, 1024]
// =====================================================================
__global__ __launch_bounds__(64) void k_attn_cat(const float* __restrict__ qkv,
                                                 float* __restrict__ out) {
    __shared__ float kb[17][64], vb[17][64], qs[64], sc[17 + KCH + 1];
    int kc = blockIdx.x, h = blockIdx.y, b = blockIdx.z, tid = threadIdx.x;
    const long tokbase = (long)b * SEQC;
    const int st3 = 3072;
    const float scale = 0.125f;

    if (kc < KCH) {
        for (int i = tid; i < 17 * 64; i += 64) {
            int m = i >> 6, d = i & 63;
            long t = tokbase + kc * LCH + m;
            kb[m][d] = qkv[t * st3 + 1024 + h * 64 + d];
            vb[m][d] = qkv[t * st3 + 2048 + h * 64 + d];
        }
        __syncthreads();
        for (int l = 0; l < LCH; ++l) {
            long qt = tokbase + kc * LCH + l;
            qs[tid] = qkv[qt * st3 + h * 64 + tid];
            __syncthreads();
            int nglb = kc;             // global keys j < kc
            int ntot = (l + 1) + nglb; // causal-local + globals
            for (int i = tid; i < ntot; i += 64) {
                const float* kv;
                if (i <= l) kv = &kb[i][0];
                else { long gt = tokbase + (long)(i - (l + 1)) * LCH;
                       kv = qkv + gt * st3 + 1024 + h * 64; }
                float s = 0.f;
                for (int d = 0; d < 64; ++d) s += qs[d] * kv[d];
                sc[i] = s * scale;
            }
            __syncthreads();
            if (tid == 0) {
                float m = -1e30f;
                for (int j = 0; j < ntot; ++j) m = fmaxf(m, sc[j]);
                float sum = 0.f;
                for (int j = 0; j < ntot; ++j) { sc[j] = __expf(sc[j] - m); sum += sc[j]; }
                float r = 1.f / sum;
                for (int j = 0; j < ntot; ++j) sc[j] *= r;
            }
            __syncthreads();
            float o = 0.f;
            for (int m = 0; m <= l; ++m) o += sc[m] * vb[m][tid];
            for (int j = 0; j < nglb; ++j) {
                long gt = tokbase + (long)j * LCH;
                o += sc[l + 1 + j] * qkv[gt * st3 + 2048 + h * 64 + tid];
            }
            out[qt * 1024 + h * 64 + tid] = o;
            __syncthreads();
        }
    } else {
        // last token: attends to 128 chunk-head keys + itself
        long qt = tokbase + SEQC - 1;
        qs[tid] = qkv[qt * st3 + h * 64 + tid];
        __syncthreads();
        const int ntot = KCH + 1;
        for (int i = tid; i < ntot; i += 64) {
            long kt = (i < KCH) ? (tokbase + (long)i * LCH) : qt;
            float s = 0.f;
            for (int d = 0; d < 64; ++d) s += qs[d] * qkv[kt * st3 + 1024 + h * 64 + d];
            sc[i] = s * scale;
        }
        __syncthreads();
        if (tid == 0) {
            float m = -1e30f;
            for (int j = 0; j < ntot; ++j) m = fmaxf(m, sc[j]);
            float sum = 0.f;
            for (int j = 0; j < ntot; ++j) { sc[j] = __expf(sc[j] - m); sum += sc[j]; }
            float r = 1.f / sum;
            for (int j = 0; j < ntot; ++j) sc[j] *= r;
        }
        __syncthreads();
        float o = 0.f;
        for (int i = 0; i < ntot; ++i) {
            long vt = (i < KCH) ? (tokbase + (long)i * LCH) : qt;
            o += sc[i] * qkv[vt * st3 + 2048 + h * 64 + tid];
        }
        out[qt * 1024 + h * 64 + tid] = o;
    }
}

// =====================================================================
// Fine input assembly: xf[b*K+k][l][512] = l==0 ? pos[k] : fwte[ids]
// =====================================================================
__global__ __launch_bounds__(256) void k_embed_fine(const int* __restrict__ ids,
                                                    const float* __restrict__ fwte,
                                                    const float* __restrict__ pos,
                                                    float* __restrict__ xf) {
    long gid = (long)blockIdx.x * blockDim.x + threadIdx.x;
    if (gid >= (long)FT_ * FD_) return;
    int d = (int)(gid % FD_);
    int l = (int)((gid / FD_) % LCH);
    int k = (int)((gid / ((long)FD_ * LCH)) % KCH);
    int b = (int)(gid / ((long)FD_ * LCH * KCH));
    float v;
    if (l == 0) v = pos[k * FD_ + d];
    else {
        int tok = ids[(b * KCH + k) * CS_ + (l - 1)];
        v = fwte[(long)tok * FD_ + d];
    }
    xf[gid] = v;
}

// =====================================================================
// Coarse input assembly
// =====================================================================
__global__ __launch_bounds__(256) void k_build_coarse(const int* __restrict__ ids,
                                                      const float* __restrict__ wte,
                                                      const float* __restrict__ dummy,
                                                      const float* __restrict__ fxb,
                                                      float* __restrict__ xc) {
    long gid = (long)blockIdx.x * blockDim.x + threadIdx.x;
    if (gid >= (long)CT_ * DIM_) return;
    int d = (int)(gid % DIM_);
    int s = (int)((gid / DIM_) % SEQC);
    int b = (int)(gid / ((long)DIM_ * SEQC));
    float v;
    if (s == SEQC - 1) v = fxb[((long)b * KCH + (KCH - 1)) * DIM_ + d];
    else {
        int k = s / LCH, l = s % LCH;
        if (l == 0) v = (k == 0) ? dummy[d] : fxb[((long)b * KCH + (k - 1)) * DIM_ + d];
        else {
            int tok = ids[(b * KCH + k) * CS_ + (l - 1)];
            v = wte[(long)tok * DIM_ + d];
        }
    }
    xc[gid] = v;
}

// =====================================================================
// Final token select + RMSNorm: block per output row (B*2048 rows)
// =====================================================================
__global__ __launch_bounds__(256) void k_select_rms(const float* __restrict__ xc,
                                                    const float* __restrict__ w,
                                                    float* __restrict__ xsel) {
    __shared__ float red[256];
    int row = blockIdx.x, tid = threadIdx.x;
    int b = row / 2048, t = row % 2048;
    int s;
    if (t < 15) s = 1 + t;
    else if (t < 2047) { int i = t - 15; s = (1 + i / 16) * LCH + (i % 16); }
    else s = SEQC - 1;
    const float* xr = xc + ((long)b * SEQC + s) * DIM_;
    float acc = 0.f;
    for (int i = tid; i < DIM_; i += 256) { float v = xr[i]; acc += v * v; }
    red[tid] = acc; __syncthreads();
    for (int o = 128; o > 0; o >>= 1) { if (tid < o) red[tid] += red[tid + o]; __syncthreads(); }
    float inv = rsqrtf(red[0] / (float)DIM_ + 1e-5f);
    for (int i = tid; i < DIM_; i += 256) xsel[(long)row * DIM_ + i] = xr[i] * inv * w[i];
}

// =====================================================================
// Host orchestration
// =====================================================================
extern "C" void kernel_launch(void* const* d_in, const int* in_sizes, int n_in,
                              void* d_out, int out_size, void* d_ws, size_t ws_size,
                              hipStream_t stream) {
    // jax pytree flattening (sorted dict keys):
    // 0 input_ids; 1 dummy_fx; 2..15 f.layers[0..1]{ln1,ln2,w1,w2,w3,wo,wqkv};
    // 16 f.norm; 17 f.pos_tokens; 18 f.proj_fx; 19 f.wte;
    // 20..47 layers[0..3]{same order}; 48 norm; 49 output; 50 wte
    const int*   ids     = (const int*)d_in[0];
    const float* dummyfx = (const float*)d_in[1];
    struct LP { const float *ln1,*ln2,*w1,*w2,*w3,*wo,*wqkv; };
    LP fl[2], cl[4];
    for (int i = 0; i < 2; ++i) {
        const int b = 2 + 7 * i;
        fl[i] = { (const float*)d_in[b], (const float*)d_in[b+1], (const float*)d_in[b+2],
                  (const float*)d_in[b+3], (const float*)d_in[b+4], (const float*)d_in[b+5],
                  (const float*)d_in[b+6] };
    }
    const float* fnorm = (const float*)d_in[16];
    const float* fpos  = (const float*)d_in[17];
    const float* projw = (const float*)d_in[18];
    const float* fwte  = (const float*)d_in[19];
    for (int i = 0; i < 4; ++i) {
        const int b = 20 + 7 * i;
        cl[i] = { (const float*)d_in[b], (const float*)d_in[b+1], (const float*)d_in[b+2],
                  (const float*)d_in[b+3], (const float*)d_in[b+4], (const float*)d_in[b+5],
                  (const float*)d_in[b+6] };
    }
    const float* cnorm = (const float*)d_in[48];
    const float* outw  = (const float*)d_in[49];
    const float* wte   = (const float*)d_in[50];

    // --- workspace bump allocator ---
    char* wsb = (char*)d_ws;
    size_t off = 0;
    auto alloc = [&](size_t bytes) -> void* {
        off = (off + 255) & ~(size_t)255;
        void* p = wsb + off;
        off += bytes;
        return p;
    };
    // transpose + convert: W[Kd][N] f32 -> WT[N][Kd] bf16
    auto convT = [&](const float* src, int Kd, int N) -> __bf16* {
        __bf16* dst = (__bf16*)alloc((size_t)Kd * N * sizeof(__bf16));
        dim3 g(N / 32, Kd / 32);
        k_cvtT<<<g, 256, 0, stream>>>(src, dst, Kd, N);
        return dst;
    };

    // --- pre-convert (and transpose) all GEMM weights to bf16 ---
    __bf16 *b_fqkv[2], *b_fwo[2], *b_fw1[2], *b_fw3[2], *b_fw2[2];
    for (int i = 0; i < 2; ++i) {
        b_fqkv[i] = convT(fl[i].wqkv, FD_, 1536);
        b_fwo[i]  = convT(fl[i].wo,   FD_, FD_);
        b_fw1[i]  = convT(fl[i].w1,   FD_, 1408);
        b_fw3[i]  = convT(fl[i].w3,   FD_, 1408);
        b_fw2[i]  = convT(fl[i].w2,   1408, FD_);
    }
    __bf16* b_proj = convT(projw, CS_ * FD_, DIM_);
    __bf16 *b_cqkv[4], *b_cwo[4], *b_cw1[4], *b_cw3[4], *b_cw2[4];
    for (int i = 0; i < 4; ++i) {
        b_cqkv[i] = convT(cl[i].wqkv, DIM_, 3072);
        b_cwo[i]  = convT(cl[i].wo,   DIM_, DIM_);
        b_cw1[i]  = convT(cl[i].w1,   DIM_, 2816);
        b_cw3[i]  = convT(cl[i].w3,   DIM_, 2816);
        b_cw2[i]  = convT(cl[i].w2,   2816, DIM_);
    }
    __bf16* b_out = convT(outw, DIM_, 32000);

    // --- activation buffers ---
    float* xf   = (float*)alloc((size_t)FT_ * FD_ * 4);
    float* rb   = (float*)alloc((size_t)CT_ * DIM_ * 4);
    float* qkvb = (float*)alloc((size_t)CT_ * 3072 * 4);
    float* attb = (float*)alloc((size_t)CT_ * DIM_ * 4);
    float* h1   = (float*)alloc((size_t)CT_ * 2816 * 4);
    float* h3   = (float*)alloc((size_t)CT_ * 2816 * 4);
    float* fxb  = (float*)alloc((size_t)BATCH_ * KCH * DIM_ * 4);
    float* xc   = (float*)alloc((size_t)CT_ * DIM_ * 4);
    float* xsel = (float*)alloc((size_t)BATCH_ * 2048 * DIM_ * 4);
    (void)ws_size; (void)in_sizes; (void)n_in; (void)out_size;

    auto gemm = [&](const float* A, long lda, const __bf16* BwT, float* C,
                    const float* Aux, int M, int N, int Kd, int mode) {
        dim3 g(N / 128, (M + 127) / 128);
        k_gemm_bf16<<<g, 256, 0, stream>>>(A, lda, BwT, C, Aux, M, N, Kd, mode);
    };

    // ================= fine transformer =================
    {
        long tot = (long)FT_ * FD_;
        k_embed_fine<<<(int)((tot + 255) / 256), 256, 0, stream>>>(ids, fwte, fpos, xf);
    }
    for (int i = 0; i < 2; ++i) {
        k_rmsnorm<<<FT_, 256, 0, stream>>>(xf, fl[i].ln1, rb, FD_);
        gemm(rb, FD_, b_fqkv[i], qkvb, nullptr, FT_, 1536, FD_, 0);
        {
            long tot = (long)FT_ * 2 * FNH_ * 32;
            k_rope<<<(int)((tot + 255) / 256), 256, 0, stream>>>(qkvb, FT_, FNH_, 0);
        }
        k_attn_full<<<BATCH_ * KCH * FNH_, 64, 0, stream>>>(qkvb, attb);
        gemm(attb, FD_, b_fwo[i], xf, xf, FT_, FD_, FD_, 1);
        k_rmsnorm<<<FT_, 256, 0, stream>>>(xf, fl[i].ln2, rb, FD_);
        gemm(rb, FD_, b_fw1[i], h1, nullptr, FT_, 1408, FD_, 0);
        gemm(rb, FD_, b_fw3[i], h3, h1, FT_, 1408, FD_, 2);
        gemm(h3, 1408, b_fw2[i], xf, xf, FT_, FD_, 1408, 1);
    }
    // fine post: rms(fnorm) then chunk projection (rows l=1..16 are contiguous)
    k_rmsnorm<<<FT_, 256, 0, stream>>>(xf, fnorm, rb, FD_);
    gemm(rb + FD_, (long)LCH * FD_, b_proj, fxb, nullptr, BATCH_ * KCH, DIM_, CS_ * FD_, 0);

    // ================= coarse transformer =================
    {
        long tot = (long)CT_ * DIM_;
        k_build_coarse<<<(int)((tot + 255) / 256), 256, 0, stream>>>(ids, wte, dummyfx, fxb, xc);
    }
    for (int i = 0; i < 4; ++i) {
        k_rmsnorm<<<CT_, 256, 0, stream>>>(xc, cl[i].ln1, rb, DIM_);
        gemm(rb, DIM_, b_cqkv[i], qkvb, nullptr, CT_, 3072, DIM_, 0);
        {
            long tot = (long)CT_ * 2 * NH_ * 32;
            k_rope<<<(int)((tot + 255) / 256), 256, 0, stream>>>(qkvb, CT_, NH_, 1);
        }
        k_attn_cat<<<dim3(KCH + 1, NH_, BATCH_), 64, 0, stream>>>(qkvb, attb);
        gemm(attb, DIM_, b_cwo[i], xc, xc, CT_, DIM_, DIM_, 1);
        k_rmsnorm<<<CT_, 256, 0, stream>>>(xc, cl[i].ln2, rb, DIM_);
        gemm(rb, DIM_, b_cw1[i], h1, nullptr, CT_, 2816, DIM_, 0);
        gemm(rb, DIM_, b_cw3[i], h3, h1, CT_, 2816, DIM_, 2);
        gemm(h3, 2816, b_cw2[i], xc, xc, CT_, DIM_, 2816, 1);
    }

    // ================= head =================
    k_select_rms<<<BATCH_ * 2048, 256, 0, stream>>>(xc, cnorm, xsel);
    gemm(xsel, DIM_, b_out, (float*)d_out, nullptr, BATCH_ * 2048, 32000, DIM_, 0);
}
